// SimpleRNN_12850542150245
// MI455X (gfx1250) — matrix operand order repaired and checked by reading, not compiled, not run on hardware
//
#include <hip/hip_runtime.h>
#include <stdint.h>

#define TSTEPS 8192
#define HDIM   1024
#define NC0    32
#define NC1    64
#define NWG    (NC0 + NC1 + 1)   // 97 persistent workgroups

typedef __attribute__((ext_vector_type(16))) __bf16    v16bf;
typedef __attribute__((ext_vector_type(8)))  float     v8f;
typedef __attribute__((ext_vector_type(8)))  uint32_t  v8u;

// Split a pair of fp32 into packed bf16 hi parts and packed bf16 lo parts.
// Packed layout: [15:0] = even element, [31:16] = odd element (K-pair in one VGPR).
__device__ __forceinline__ void split_pair(float e, float o, uint32_t& hi, uint32_t& lo) {
  uint32_t be = __float_as_uint(e), bo = __float_as_uint(o);
  hi = (bo & 0xFFFF0000u) | (be >> 16);
  float el = e - __uint_as_float(be & 0xFFFF0000u);
  float ol = o - __uint_as_float(bo & 0xFFFF0000u);
  lo = (__float_as_uint(ol) & 0xFFFF0000u) | (__float_as_uint(el) >> 16);
}

__device__ __forceinline__ v16bf mkfrag(uint4 a, uint4 b) {
  v8u u = {a.x, a.y, a.z, a.w, b.x, b.y, b.z, b.w};
  return __builtin_bit_cast(v16bf, u);
}

__device__ __forceinline__ v8f wmma_bf16(v16bf a, v16bf b, v8f c) {
  // D = A(16x32 bf16) * B(32x16 bf16) + C(f32), fp32 accumulate
  return __builtin_amdgcn_wmma_f32_16x16x32_bf16(false, a, false, b, (short)0, c, false, false);
}

__device__ __forceinline__ float sigf(float x) { return 1.0f / (1.0f + __expf(-x)); }

// ---------------------------------------------------------------------------
// Zero the h double-buffers and the grid-barrier counter (runs every call).
__global__ void init_state(float* h0b, float* h1b, uint32_t* counter) {
  for (int k = threadIdx.x; k < 2 * HDIM; k += blockDim.x) { h0b[k] = 0.f; h1b[k] = 0.f; }
  if (threadIdx.x == 0) *counter = 0u;
}

// ---------------------------------------------------------------------------
// u0 = W_ih0 @ w_in ; vb0 = W_ih0 @ b_in + b_ih0 + b_hh0 ; b1s = b_ih1 + b_hh1
__global__ __launch_bounds__(256)
void prep_vectors(const float* __restrict__ W_in, const float* __restrict__ b_in,
                  const float* __restrict__ W_ih0,
                  const float* __restrict__ b_ih0, const float* __restrict__ b_hh0,
                  const float* __restrict__ b_ih1, const float* __restrict__ b_hh1,
                  float* __restrict__ u0, float* __restrict__ vb0, float* __restrict__ b1s) {
  int row  = (blockIdx.x * blockDim.x + threadIdx.x) >> 5;
  int lane = threadIdx.x & 31;
  if (row >= 4 * HDIM) return;
  const float* wr = W_ih0 + (size_t)row * HDIM;
  float su = 0.f, sv = 0.f;
  for (int k = lane; k < HDIM; k += 32) {
    float w = wr[k];
    su += w * W_in[k];
    sv += w * b_in[k];
  }
  for (int off = 16; off > 0; off >>= 1) {
    su += __shfl_down(su, off, 32);
    sv += __shfl_down(sv, off, 32);
  }
  if (lane == 0) {
    u0[row]  = su;
    vb0[row] = sv + b_ih0[row] + b_hh0[row];
    b1s[row] = b_ih1[row] + b_hh1[row];
  }
}

// ---------------------------------------------------------------------------
// Pre-pack W_hh0, W_ih1, W_hh1 into WMMA B-fragment layout (hi/lo bf16 split).
// Fragment (g = gate-row-block of 16, c = 32-wide K chunk): per lane 64 bytes,
// [0..31] = hi-packed (8 dwords), [32..63] = lo-packed.
// B lane l: N = l&15, K = (l&16) + 0..15 of the chunk; VGPR j = K pair (2j,2j+1).
__global__ __launch_bounds__(256)
void pack_weights(const float* __restrict__ Whh0, const float* __restrict__ Wih1,
                  const float* __restrict__ Whh1, uint32_t* __restrict__ wpack) {
  int wave = threadIdx.x >> 5, lane = threadIdx.x & 31;
  int fi = blockIdx.x * 8 + wave;          // 0 .. 3*8192-1 fragments
  int m  = fi >> 13;
  int rem = fi & 8191;
  int g = rem >> 5, c = rem & 31;
  const float* W = (m == 0) ? Whh0 : ((m == 1) ? Wih1 : Whh1);
  int n    = lane & 15;
  int koff = lane & 16;
  int row  = (g >> 6) * HDIM + (g & 63) * 16 + n;   // gate*1024 + unitgroup*16 + n
  int col  = c * 32 + koff;
  const float4* src = (const float4*)&W[(size_t)row * HDIM + col];
  float4 f0 = src[0], f1 = src[1], f2 = src[2], f3 = src[3];
  uint32_t hi[8], lo[8];
  split_pair(f0.x, f0.y, hi[0], lo[0]);
  split_pair(f0.z, f0.w, hi[1], lo[1]);
  split_pair(f1.x, f1.y, hi[2], lo[2]);
  split_pair(f1.z, f1.w, hi[3], lo[3]);
  split_pair(f2.x, f2.y, hi[4], lo[4]);
  split_pair(f2.z, f2.w, hi[5], lo[5]);
  split_pair(f3.x, f3.y, hi[6], lo[6]);
  split_pair(f3.z, f3.w, hi[7], lo[7]);
  uint32_t* dst = wpack + (size_t)fi * 512 + (size_t)lane * 16;
  ((uint4*)dst)[0] = make_uint4(hi[0], hi[1], hi[2], hi[3]);
  ((uint4*)dst)[1] = make_uint4(hi[4], hi[5], hi[6], hi[7]);
  ((uint4*)dst)[2] = make_uint4(lo[0], lo[1], lo[2], lo[3]);
  ((uint4*)dst)[3] = make_uint4(lo[4], lo[5], lo[6], lo[7]);
}

// ---------------------------------------------------------------------------
// Persistent pipelined LSTM. Round r (1..T+2):
//   cell0 (r<=T):        h0(r)   = LSTM0(x(r), h0(r-1))          [c0 in regs]
//   cell1 (2<=r<=T+1):   h1(r-1) = LSTM1(h0(r-1), h1(r-2))       [c1 in regs]
//   out   (r>=3):        y(r-2)  = W_out . h1(r-2) + b_out
// One grid barrier per round.
__global__ __launch_bounds__(256, 1)
void lstm_persistent(const float* __restrict__ inputs,
                     const float* __restrict__ W_out, const float* __restrict__ b_out,
                     const uint32_t* __restrict__ wpack,
                     const float* __restrict__ u0, const float* __restrict__ vb0,
                     const float* __restrict__ b1s,
                     float* __restrict__ h0b, float* __restrict__ h1b,
                     uint32_t* __restrict__ counter,
                     float* __restrict__ out) {
  __shared__ uint32_t hpA_hi[512], hpA_lo[512], hpB_hi[512], hpB_lo[512];
  __shared__ float    gl[8][16];

  const int wg   = blockIdx.x;
  const int tid  = threadIdx.x;
  const int wave = tid >> 5;
  const int lane = tid & 31;

  const bool isC0  = (wg < NC0);
  const bool isC1  = (wg >= NC0 && wg < NC0 + NC1);
  const bool isOut = (wg == NC0 + NC1);

  int m = 0, gblock = 0;
  bool useB = false;
  if (isC0) {                       // 2 unit-groups, 4 gate rows each, all W_hh0
    int ug = wg * 2 + (wave >> 2);
    m = 0; gblock = (wave & 3) * 64 + ug;
  } else if (isC1) {                // waves 0-3: W_ih1 (A=h0), waves 4-7: W_hh1 (A=h1)
    int ug = wg - NC0;
    m = (wave < 4) ? 1 : 2;
    gblock = (wave & 3) * 64 + ug;
    useB = (wave >= 4);
  }
  const uint32_t* wbase = wpack + ((size_t)m * 8192 + (size_t)gblock * 32) * 512
                                + (size_t)lane * 16;

  float c_state = 0.0f;                       // per-lane resident cell state
  const int a_dw = (lane < 16) ? 0 : 8;       // A-fragment K sub-offset per lane half

  for (int r = 1; r <= TSTEPS + 2; ++r) {
    const int p = r & 1;
    const float* h0_rd = h0b + (p ^ 1) * HDIM;   // h0(r-1)
    float*       h0_wr = h0b + p * HDIM;         // h0(r)
    const float* h1_rd = h1b + p * HDIM;         // h1(r-2)
    float*       h1_wr = h1b + (p ^ 1) * HDIM;   // h1(r-1)

    const bool act0 = isC0 && (r <= TSTEPS);
    const bool act1 = isC1 && (r >= 2) && (r <= TSTEPS + 1);

    // Stage packed bf16 hi/lo copies of the needed h vectors in LDS.
    if (act0 || act1) {
      float4 fa = *(const float4*)&h0_rd[tid * 4];
      uint32_t ph0, ph1, pl0, pl1;
      split_pair(fa.x, fa.y, ph0, pl0);
      split_pair(fa.z, fa.w, ph1, pl1);
      hpA_hi[tid * 2] = ph0; hpA_hi[tid * 2 + 1] = ph1;
      hpA_lo[tid * 2] = pl0; hpA_lo[tid * 2 + 1] = pl1;
      if (isC1) {
        float4 fb = *(const float4*)&h1_rd[tid * 4];
        split_pair(fb.x, fb.y, ph0, pl0);
        split_pair(fb.z, fb.w, ph1, pl1);
        hpB_hi[tid * 2] = ph0; hpB_hi[tid * 2 + 1] = ph1;
        hpB_lo[tid * 2] = pl0; hpB_lo[tid * 2 + 1] = pl1;
      }
    }
    __syncthreads();

    // GEMV tile-row: 32 K-chunks, 3 bf16 WMMAs per chunk (fp32-accurate split).
    if (act0 || act1) {
      const uint32_t* hph = useB ? hpB_hi : hpA_hi;
      const uint32_t* hpl = useB ? hpB_lo : hpA_lo;
      v8f acc0 = {0.f, 0.f, 0.f, 0.f, 0.f, 0.f, 0.f, 0.f};
      v8f acc1 = {0.f, 0.f, 0.f, 0.f, 0.f, 0.f, 0.f, 0.f};
#pragma unroll 2
      for (int c = 0; c < 32; ++c) {
        const uint4* gp = (const uint4*)(wbase + (size_t)c * 512);
        if (c + 1 < 32) __builtin_prefetch((const void*)(wbase + (size_t)(c + 1) * 512), 0, 3);
        uint4 bh0 = gp[0], bh1 = gp[1], bl0 = gp[2], bl1 = gp[3];
        int ad = c * 16 + a_dw;
        uint4 ah0 = *(const uint4*)&hph[ad];
        uint4 ah1 = *(const uint4*)&hph[ad + 4];
        uint4 al0 = *(const uint4*)&hpl[ad];
        uint4 al1 = *(const uint4*)&hpl[ad + 4];
        v16bf A_hi = mkfrag(ah0, ah1);
        v16bf A_lo = mkfrag(al0, al1);
        v16bf B_hi = mkfrag(bh0, bh1);
        v16bf B_lo = mkfrag(bl0, bl1);
        acc0 = wmma_bf16(A_hi, B_hi, acc0);
        acc1 = wmma_bf16(A_hi, B_lo, acc1);
        acc1 = wmma_bf16(A_lo, B_hi, acc1);
      }
      if (lane < 16) gl[wave][lane] = acc0[0] + acc1[0];   // D row M=0: 16 gate dots
    }
    __syncthreads();

    // Cell updates (c-state stays in registers for the whole sequence).
    if (act0 && tid < 32) {
      int half = tid >> 4, j16 = tid & 15;
      int j = (wg * 2 + half) * 16 + j16;
      float a_t = inputs[r - 1];
      float gi = gl[half * 4 + 0][j16] + a_t * u0[j]            + vb0[j];
      float gf = gl[half * 4 + 1][j16] + a_t * u0[HDIM + j]     + vb0[HDIM + j];
      float gg = gl[half * 4 + 2][j16] + a_t * u0[2 * HDIM + j] + vb0[2 * HDIM + j];
      float go = gl[half * 4 + 3][j16] + a_t * u0[3 * HDIM + j] + vb0[3 * HDIM + j];
      c_state = sigf(gf) * c_state + sigf(gi) * tanhf(gg);
      h0_wr[j] = sigf(go) * tanhf(c_state);
    }
    if (act1 && tid < 16) {
      int j = (wg - NC0) * 16 + tid;
      float gi = gl[0][tid] + gl[4][tid] + b1s[j];
      float gf = gl[1][tid] + gl[5][tid] + b1s[HDIM + j];
      float gg = gl[2][tid] + gl[6][tid] + b1s[2 * HDIM + j];
      float go = gl[3][tid] + gl[7][tid] + b1s[3 * HDIM + j];
      c_state = sigf(gf) * c_state + sigf(gi) * tanhf(gg);
      h1_wr[j] = sigf(go) * tanhf(c_state);
    }
    if (isOut && (r >= 3) && wave == 0) {
      float s = 0.f;
      for (int k = lane; k < HDIM; k += 32) s += W_out[k] * h1_rd[k];
      for (int off = 16; off > 0; off >>= 1) s += __shfl_down(s, off, 32);
      if (lane == 0) out[r - 3] = s + b_out[0];
    }

    // Device-wide barrier (release stores, acquire loads).
    __threadfence();
    __syncthreads();
    if (tid == 0) {
      __hip_atomic_fetch_add(counter, 1u, __ATOMIC_RELEASE, __HIP_MEMORY_SCOPE_AGENT);
      const uint32_t tgt = (uint32_t)NWG * (uint32_t)r;
      while (__hip_atomic_load(counter, __ATOMIC_ACQUIRE, __HIP_MEMORY_SCOPE_AGENT) < tgt)
        __builtin_amdgcn_s_sleep(2);
    }
    __syncthreads();
    __threadfence();
  }
}

// ---------------------------------------------------------------------------
extern "C" void kernel_launch(void* const* d_in, const int* in_sizes, int n_in,
                              void* d_out, int out_size, void* d_ws, size_t ws_size,
                              hipStream_t stream) {
  (void)in_sizes; (void)n_in; (void)out_size; (void)ws_size;
  const float* inputs = (const float*)d_in[0];
  const float* W_in   = (const float*)d_in[1];
  const float* b_in   = (const float*)d_in[2];
  const float* W_ih0  = (const float*)d_in[3];
  const float* W_hh0  = (const float*)d_in[4];
  const float* b_ih0  = (const float*)d_in[5];
  const float* b_hh0  = (const float*)d_in[6];
  const float* W_ih1  = (const float*)d_in[7];
  const float* W_hh1  = (const float*)d_in[8];
  const float* b_ih1  = (const float*)d_in[9];
  const float* b_hh1  = (const float*)d_in[10];
  const float* W_out  = (const float*)d_in[11];
  const float* b_out  = (const float*)d_in[12];

  // Workspace layout (4-byte units): [packed weights 48MB][u0][vb0][b1s][h0 x2][h1 x2][counter]
  uint32_t* wpack = (uint32_t*)d_ws;
  const size_t WPACK_U32 = (size_t)3 * 8192 * 512;
  float* fws = (float*)d_ws;
  float* u0  = fws + WPACK_U32;
  float* vb0 = u0 + 4 * HDIM;
  float* b1s = vb0 + 4 * HDIM;
  float* h0b = b1s + 4 * HDIM;
  float* h1b = h0b + 2 * HDIM;
  uint32_t* counter = (uint32_t*)(h1b + 2 * HDIM);

  hipLaunchKernelGGL(init_state, dim3(1), dim3(256), 0, stream, h0b, h1b, counter);
  hipLaunchKernelGGL(prep_vectors, dim3(512), dim3(256), 0, stream,
                     W_in, b_in, W_ih0, b_ih0, b_hh0, b_ih1, b_hh1, u0, vb0, b1s);
  hipLaunchKernelGGL(pack_weights, dim3(3072), dim3(256), 0, stream,
                     W_hh0, W_ih1, W_hh1, wpack);
  hipLaunchKernelGGL(lstm_persistent, dim3(NWG), dim3(256), 0, stream,
                     inputs, W_out, b_out, wpack, u0, vb0, b1s, h0b, h1b, counter,
                     (float*)d_out);
}
